// Kalman_VIO_9861244911894
// MI455X (gfx1250) — matrix operand order repaired and checked by reading, not compiled, not run on hardware
//
#include <hip/hip_runtime.h>
#include <stdint.h>

#define BATCH 2048
#define TT    10
#define WN    11
#define HID   256

typedef __attribute__((ext_vector_type(16))) _Float16 v16h;
typedef __attribute__((ext_vector_type(8)))  _Float16 v8h;
typedef __attribute__((ext_vector_type(8)))  float    v8f;

union Frag { v16h v; v8h p[2]; };

// ---- WMMA fragment loads (per CDNA5 ISA 7.12.2 layouts) --------------------
// A (16x32 f16): lane half h, VGPR0-3 = K 8h..8h+7, VGPR4-7 = K 16+8h..23+8h
__device__ inline v16h ldA(const _Float16* row, int kb, int half) {
  Frag f;
  f.p[0] = *(const v8h*)(row + kb + 8 * half);
  f.p[1] = *(const v8h*)(row + kb + 16 + 8 * half);
  return f.v;
}
// B (32x16 f16), stored as W row-major (N x K): lane half h, VGPR0-3 = K 8h..8h+7, VGPR4-7 = K 16+8h..23+8h
__device__ inline v16h ldB(const _Float16* row, int kb, int half) {
  Frag f;
  f.p[0] = *(const v8h*)(row + kb + 8 * half);
  f.p[1] = *(const v8h*)(row + kb + 16 + 8 * half);
  return f.v;
}
__device__ inline v8f wmma32(v16h a, v16h b, v8f c) {
  return __builtin_amdgcn_wmma_f32_16x16x32_f16(false, a, false, b, (short)0, c,
                                                false, false);
}
__device__ inline float sigm(float x) { return 1.0f / (1.0f + __expf(-x)); }

// ---- Generic GEMM: out = act(A[MxK] * W[N16xK]^T + bias) -------------------
__global__ void gemm16(const _Float16* __restrict__ A, int lda,
                       const _Float16* __restrict__ Wt, int ldk,
                       const float* __restrict__ bias, int K, int tilesN,
                       int totalTiles, _Float16* outH, int ldnH, float* outF,
                       int ldnF, int relu) {
  int wave = threadIdx.x >> 5;
  int lane = threadIdx.x & 31;
  int half = lane >> 4;
  int r = lane & 15;
  int tile = blockIdx.x * 8 + wave;
  if (tile >= totalTiles) return;  // wave-uniform
  int ti = tile / tilesN, tj = tile % tilesN;
  const _Float16* arow = A + (size_t)(ti * 16 + r) * lda;
  const _Float16* wrow = Wt + (size_t)(tj * 16 + r) * ldk;
  v8f acc = {};
  for (int kb = 0; kb < K; kb += 32)
    acc = wmma32(ldA(arow, kb, half), ldB(wrow, kb, half), acc);
#pragma unroll
  for (int v = 0; v < 8; v++) {
    int m = ti * 16 + v + 8 * half;
    int n = tj * 16 + r;
    float val = acc[v] + bias[n];
    if (relu) val = fmaxf(val, 0.0f);
    if (outH) outH[(size_t)m * ldnH + n] = (_Float16)val;
    if (outF) outF[(size_t)m * ldnF + n] = val;
  }
}

// ---- helper: accumulate 8 n-tiles (one wave) of A(16xK) * W(N x K)^T -------
__device__ inline void accum_part(v8f acc[8], const _Float16* arow,
                                  const _Float16* Wg, int ldk, int K, int wave,
                                  int half, int r) {
  const _Float16* wrow[8];
#pragma unroll
  for (int j = 0; j < 8; j++)
    wrow[j] = Wg + (size_t)((wave * 8 + j) * 16 + r) * ldk;
  for (int kb = 0; kb < K; kb += 32) {
    v16h a = ldA(arow, kb, half);
#pragma unroll
    for (int j = 0; j < 8; j++) {
      v16h b = ldB(wrow[j], kb, half);
      acc[j] = wmma32(a, b, acc[j]);
    }
  }
}

// ---- Fused 2-layer LSTM over all 11 window steps ---------------------------
// block = 256 threads (8 waves) handles 16 batch rows; h/c/gates in LDS.
#define SM_GS 65536  // gates 16x1024 f32
#define SM_C0 16384
#define SM_C1 16384
#define SM_H0 8192
#define SM_H1 8192
#define SM_XS 11264  // x tile 11x16x32 f16
#define LSTM_SMEM (SM_GS + SM_C0 + SM_C1 + SM_H0 + SM_H1 + SM_XS)

__global__ void lstm2(const float* __restrict__ accp,
                      const float* __restrict__ angp, int t, int imuMode,
                      const _Float16* __restrict__ wih0,
                      const _Float16* __restrict__ whh0,
                      const _Float16* __restrict__ wih1,
                      const _Float16* __restrict__ whh1,
                      const float* __restrict__ bsum0,
                      const float* __restrict__ bsum1, _Float16* featOut) {
  extern __shared__ uint8_t smem[];
  float* gs = (float*)smem;
  float* c0 = (float*)(smem + SM_GS);
  float* c1 = (float*)(smem + SM_GS + SM_C0);
  _Float16* h0 = (_Float16*)(smem + SM_GS + SM_C0 + SM_C1);
  _Float16* h1 = (_Float16*)(smem + SM_GS + SM_C0 + SM_C1 + SM_H0);
  _Float16* xs = (_Float16*)(smem + SM_GS + SM_C0 + SM_C1 + SM_H0 + SM_H1);

  int tid = threadIdx.x;
  int wave = tid >> 5, lane = tid & 31, half = lane >> 4, r = lane & 15;

  // load + convert input window (padded to K=32)
  for (int i = tid; i < WN * 16 * 32; i += 256) {
    int s = i / 512, rem = i % 512, row = rem >> 5, col = rem & 31;
    size_t ib = ((size_t)(blockIdx.x * 16 + row) * TT + t) * WN + s;
    float v = 0.0f;
    if (imuMode) {
      if (col < 3) v = accp[ib * 3 + col];
      else if (col < 6) v = angp[ib * 3 + (col - 3)];
    } else {
      if (col < 3) v = angp[ib * 3 + col];
    }
    xs[i] = (_Float16)v;
  }
  for (int i = tid; i < 16 * HID; i += 256) {
    c0[i] = 0.0f; c1[i] = 0.0f;
    h0[i] = (_Float16)0.0f; h1[i] = (_Float16)0.0f;
  }
  __syncthreads();

  for (int s = 0; s < WN; s++) {
    // ---- layer 0 gates = x_s @ Wih0^T + h0 @ Whh0^T + b
    {
      v8f acc[8];
#pragma unroll
      for (int j = 0; j < 8; j++) acc[j] = (v8f){};
      accum_part(acc, xs + (size_t)(s * 16 + r) * 32, wih0, 32, 32, wave, half, r);
      accum_part(acc, h0 + (size_t)r * HID, whh0, HID, HID, wave, half, r);
#pragma unroll
      for (int j = 0; j < 8; j++) {
        int n0 = (wave * 8 + j) * 16 + r;
#pragma unroll
        for (int v = 0; v < 8; v++)
          gs[(v + 8 * half) * 1024 + n0] = acc[j][v] + bsum0[n0];
      }
    }
    __syncthreads();
    for (int i = tid; i < 16 * HID; i += 256) {
      int row = i >> 8, j = i & 255;
      float gi = gs[row * 1024 + j], gf = gs[row * 1024 + j + 256];
      float gg = gs[row * 1024 + j + 512], go = gs[row * 1024 + j + 768];
      float c = sigm(gf) * c0[i] + sigm(gi) * tanhf(gg);
      c0[i] = c;
      h0[i] = (_Float16)(sigm(go) * tanhf(c));
    }
    __syncthreads();
    // ---- layer 1 gates = h0 @ Wih1^T + h1 @ Whh1^T + b
    {
      v8f acc[8];
#pragma unroll
      for (int j = 0; j < 8; j++) acc[j] = (v8f){};
      accum_part(acc, h0 + (size_t)r * HID, wih1, HID, HID, wave, half, r);
      accum_part(acc, h1 + (size_t)r * HID, whh1, HID, HID, wave, half, r);
#pragma unroll
      for (int j = 0; j < 8; j++) {
        int n0 = (wave * 8 + j) * 16 + r;
#pragma unroll
        for (int v = 0; v < 8; v++)
          gs[(v + 8 * half) * 1024 + n0] = acc[j][v] + bsum1[n0];
      }
    }
    __syncthreads();
    for (int i = tid; i < 16 * HID; i += 256) {
      int row = i >> 8, j = i & 255;
      float gi = gs[row * 1024 + j], gf = gs[row * 1024 + j + 256];
      float gg = gs[row * 1024 + j + 512], go = gs[row * 1024 + j + 768];
      float c = sigm(gf) * c1[i] + sigm(gi) * tanhf(gg);
      c1[i] = c;
      h1[i] = (_Float16)(sigm(go) * tanhf(c));
    }
    __syncthreads();
  }
  // write last hidden into feature buffer columns [256,512)
  for (int i = tid; i < 16 * HID; i += 256) {
    int row = i >> 8, j = i & 255;
    size_t b = (size_t)blockIdx.x * 16 + row;
    featOut[b * 512 + 256 + j] = h1[i];
  }
}

// ---- small utility kernels -------------------------------------------------
__global__ void cvt_pad(const float* __restrict__ src, _Float16* dst, int N,
                        int K, int Np, int Kp) {
  int i = blockIdx.x * blockDim.x + threadIdx.x;
  if (i >= Np * Kp) return;
  int row = i / Kp, col = i % Kp;
  dst[i] = (row < N && col < K) ? (_Float16)src[(size_t)row * K + col]
                                : (_Float16)0.0f;
}
__global__ void bias_pad(const float* __restrict__ src, float* dst, int N,
                         int Np) {
  int i = blockIdx.x * blockDim.x + threadIdx.x;
  if (i >= Np) return;
  dst[i] = (i < N) ? src[i] : 0.0f;
}
__global__ void bias_sum(const float* __restrict__ a,
                         const float* __restrict__ b, float* dst, int n) {
  int i = blockIdx.x * blockDim.x + threadIdx.x;
  if (i < n) dst[i] = a[i] + b[i];
}
__global__ void pack_xt_k(const float* __restrict__ xt, _Float16* dst) {
  int i = blockIdx.x * blockDim.x + threadIdx.x;
  if (i >= BATCH * 32) return;
  int b = i >> 5, c = i & 31;
  dst[i] = (c < 6) ? (_Float16)xt[b * 6 + c] : (_Float16)0.0f;
}
__global__ void copy_x0(const _Float16* __restrict__ ft, _Float16* fr) {
  int i = blockIdx.x * blockDim.x + threadIdx.x;
  if (i >= BATCH * 256) return;
  int b = i >> 8, j = i & 255;
  fr[(size_t)b * 512 + j] = ft[(size_t)b * 512 + j];
}
__global__ void init_state(float* xt, float* P) {
  int i = blockIdx.x * blockDim.x + threadIdx.x;
  if (i < BATCH * 36) {
    int rr = (i % 36) / 6, cc = i % 6;
    P[i] = (rr == cc) ? 1.0f : 0.0f;
  }
  if (i < BATCH * 6) xt[i] = 0.0f;
}

// ---- Kalman combine + update (one thread per batch element) ----------------
#define O_TRANS 0ull
#define O_XT    122880ull
#define O_P     245760ull
#define O_Q     983040ull
#define O_A     1720320ull
#define O_K     2457600ull
#define O_AXT   3194880ull

__global__ void kalman(int t, const float* __restrict__ ho,
                       const float* __restrict__ cw,
                       const float* __restrict__ zt,
                       const float* __restrict__ Rm,
                       const float* __restrict__ accp,
                       const float* __restrict__ angp, float* xt_prev,
                       float* P_prev, float* out) {
  int b = blockIdx.x * blockDim.x + threadIdx.x;
  if (b >= BATCH) return;
  const int RA[12] = {0, 1, 2, 3, 4, 5, 1, 1, 3, 3, 5, 5};
  const int CA[12] = {0, 1, 2, 3, 4, 5, 3, 5, 1, 5, 1, 3};
  size_t bt = (size_t)b * TT + t;
  // branch memory order: 0=left 1=right 2=straight; weights: 0=L,1=S,2=R
  float wb[3];
  wb[0] = cw[bt * 3 + 0]; wb[1] = cw[bt * 3 + 2]; wb[2] = cw[bt * 3 + 1];
  const float* TRA[3]; const float* ROT[3]; const float* AH[3]; const float* QH[3];
  for (int br = 0; br < 3; br++) {
    TRA[br] = ho + ((size_t)(0 * 3 + br) * BATCH + b) * 16;
    ROT[br] = ho + ((size_t)(1 * 3 + br) * BATCH + b) * 16;
    AH[br]  = ho + ((size_t)(2 * 3 + br) * BATCH + b) * 16;
    QH[br]  = ho + ((size_t)(3 * 3 + br) * BATCH + b) * 16;
  }
  float xtp[6] = {0, 0, 0, 0, 0, 0};
  for (int br = 0; br < 3; br++) {
    float w = wb[br];
    xtp[0] += w * ROT[br][0]; xtp[1] += w * ROT[br][1]; xtp[2] += w * ROT[br][2];
    xtp[3] += w * TRA[br][0]; xtp[4] += w * TRA[br][1]; xtp[5] += w * TRA[br][2];
  }
  float A[36]; float Qm[36];
  for (int i = 0; i < 36; i++) { A[i] = 0.0f; Qm[i] = 0.0f; }
  for (int k = 0; k < 12; k++) {
    float s = wb[0] * AH[0][k] + wb[1] * AH[1][k] + wb[2] * AH[2][k];
    A[RA[k] * 6 + CA[k]] = s;
  }
  for (int br = 0; br < 3; br++) {
    const float* q = QH[br];
    float w = wb[br];
    for (int i = 0; i < 6; i++) Qm[i * 6 + i] += w * q[i] * q[i];
    float o35 = q[6] * q[3] * q[5];
    float o13 = q[7] * q[3] * q[1];
    float o15 = q[8] * q[5] * q[1];
    Qm[3 * 6 + 5] += w * o35; Qm[5 * 6 + 3] += w * o35;
    Qm[3 * 6 + 1] += w * o13; Qm[1 * 6 + 3] += w * o13;
    Qm[1 * 6 + 5] += w * o15; Qm[5 * 6 + 1] += w * o15;
  }
  float xo[6], P[36];
  for (int i = 0; i < 6; i++) xo[i] = xt_prev[(size_t)b * 6 + i];
  for (int i = 0; i < 36; i++) P[i] = P_prev[(size_t)b * 36 + i];
  float axt[6];
  for (int i = 0; i < 6; i++) {
    float s = 0.0f;
    for (int k = 0; k < 6; k++) s += A[i * 6 + k] * xo[k];
    axt[i] = s;
  }
  float AP[36], Pp[36];
  for (int i = 0; i < 6; i++)
    for (int j = 0; j < 6; j++) {
      float s = 0.0f;
      for (int k = 0; k < 6; k++) s += A[i * 6 + k] * P[k * 6 + j];
      AP[i * 6 + j] = s;
    }
  for (int i = 0; i < 6; i++)
    for (int j = 0; j < 6; j++) {
      float s = 0.0f;
      for (int k = 0; k < 6; k++) s += AP[i * 6 + k] * A[j * 6 + k];
      Pp[i * 6 + j] = s + Qm[i * 6 + j];
    }
  // S = Pp + R; solve S^T X = Pp^T; K = X^T  (matches jnp.linalg.solve form)
  float M[6][12];
  for (int i = 0; i < 6; i++)
    for (int j = 0; j < 6; j++) {
      float S_ji = Pp[j * 6 + i] + Rm[bt * 36 + j * 6 + i];
      M[i][j] = S_ji;          // S^T
      M[i][6 + j] = Pp[j * 6 + i];  // Pp^T
    }
  for (int col = 0; col < 6; col++) {
    int piv = col; float best = fabsf(M[col][col]);
    for (int r2 = col + 1; r2 < 6; r2++) {
      float v = fabsf(M[r2][col]);
      if (v > best) { best = v; piv = r2; }
    }
    if (piv != col)
      for (int j2 = 0; j2 < 12; j2++) {
        float tv = M[col][j2]; M[col][j2] = M[piv][j2]; M[piv][j2] = tv;
      }
    float inv = 1.0f / M[col][col];
    for (int j2 = 0; j2 < 12; j2++) M[col][j2] *= inv;
    for (int r2 = 0; r2 < 6; r2++) {
      if (r2 == col) continue;
      float f = M[r2][col];
      for (int j2 = 0; j2 < 12; j2++) M[r2][j2] -= f * M[col][j2];
    }
  }
  float Km[36];
  for (int i = 0; i < 6; i++)
    for (int j = 0; j < 6; j++) Km[i * 6 + j] = M[j][6 + i];
  // stationary override
  {
    size_t f0 = (bt * WN) * 3, lw = (bt * WN + (WN - 1)) * 3;
    float av = (fabsf(accp[lw] - accp[f0]) + fabsf(accp[lw + 1] - accp[f0 + 1]) +
                fabsf(accp[lw + 2] - accp[f0 + 2])) * (1.0f / 3.0f);
    float gv = (fabsf(angp[lw] - angp[f0]) + fabsf(angp[lw + 1] - angp[f0 + 1]) +
                fabsf(angp[lw + 2] - angp[f0 + 2])) * (1.0f / 3.0f);
    if (av < 0.01f && gv < 0.01f)
      for (int i = 0; i < 6; i++)
        for (int j = 0; j < 6; j++) Km[i * 6 + j] = (i == j) ? 1.0f : 0.0f;
  }
  float innov[6], xn[6];
  for (int i = 0; i < 6; i++) innov[i] = zt[bt * 6 + i] - xtp[i];
  for (int i = 0; i < 6; i++) {
    float s = xtp[i];
    for (int j = 0; j < 6; j++) s += Km[i * 6 + j] * innov[j];
    xn[i] = s;
  }
  float Pn[36];
  for (int i = 0; i < 6; i++)
    for (int j = 0; j < 6; j++) {
      float s = Pp[i * 6 + j];
      for (int k = 0; k < 6; k++) s -= Km[i * 6 + k] * Pp[k * 6 + j];
      Pn[i * 6 + j] = s;
    }
  size_t bt6 = bt * 6, bt36 = bt * 36;
  for (int i = 0; i < 6; i++) {
    out[O_TRANS + bt6 + i] = xtp[i];
    out[O_XT + bt6 + i] = xn[i];
    out[O_AXT + bt6 + i] = axt[i];
    xt_prev[(size_t)b * 6 + i] = xn[i];
  }
  for (int i = 0; i < 36; i++) {
    out[O_P + bt36 + i] = Pn[i];
    out[O_Q + bt36 + i] = Qm[i];
    out[O_A + bt36 + i] = A[i];
    out[O_K + bt36 + i] = Km[i];
    P_prev[(size_t)b * 36 + i] = Pn[i];
  }
}

// ---- host ------------------------------------------------------------------
static inline void launch_gemm(hipStream_t s, const _Float16* A, int lda,
                               const _Float16* W, int ldk, const float* bias,
                               int M, int K, int N16, _Float16* oh, int ldh,
                               float* of, int ldf, int relu) {
  int tilesN = N16 / 16;
  int total = (M / 16) * tilesN;
  int blocks = (total + 7) / 8;
  gemm16<<<blocks, 256, 0, s>>>(A, lda, W, ldk, bias, K, tilesN, total, oh, ldh,
                                of, ldf, relu);
}

extern "C" void kernel_launch(void* const* d_in, const int* in_sizes, int n_in,
                              void* d_out, int out_size, void* d_ws,
                              size_t ws_size, hipStream_t stream) {
  (void)in_sizes; (void)n_in; (void)out_size; (void)ws_size;
  // Input index map (top-level dict insertion order; params dict key-sorted):
  const float* zt = (const float*)d_in[0];
  const float* Rm = (const float*)d_in[1];
  const float* cw = (const float*)d_in[2];
  const float* acc = (const float*)d_in[3];
  const float* ang = (const float*)d_in[4];
  // head types: 0=t(tra,feat_t) 1=r(rot,feat_r) 2=A(feat_t) 3=Q(feat_t)
  const int typeBase[4][3] = {{81, 87, 93}, {63, 69, 75}, {5, 11, 17}, {23, 29, 35}};
  const int headN[4] = {3, 3, 12, 9};
  const int ANG_BASE = 41, IMU_BASE = 49, INIT_BASE = 57;

  uint8_t* base = (uint8_t*)d_ws;
  size_t off = 0;
  auto carve = [&](size_t bytes) -> void* {
    void* p = base + off;
    off += (bytes + 255) & ~(size_t)255;
    return p;
  };
  // f16 weights
  _Float16* w_init0 = (_Float16*)carve(128 * 32 * 2);
  _Float16* w_init1 = (_Float16*)carve(256 * 128 * 2);
  _Float16* w_init2 = (_Float16*)carve(256 * 256 * 2);
  _Float16* imu_w[4]; // wih0, whh0, wih1, whh1
  imu_w[0] = (_Float16*)carve(1024 * 32 * 2);
  for (int i = 1; i < 4; i++) imu_w[i] = (_Float16*)carve(1024 * 256 * 2);
  _Float16* ang_w[4];
  ang_w[0] = (_Float16*)carve(1024 * 32 * 2);
  for (int i = 1; i < 4; i++) ang_w[i] = (_Float16*)carve(1024 * 256 * 2);
  _Float16* hw0[12]; _Float16* hw1[12]; _Float16* hw2[12];
  for (int h = 0; h < 12; h++) {
    hw0[h] = (_Float16*)carve(256 * 512 * 2);
    hw1[h] = (_Float16*)carve(128 * 256 * 2);
    hw2[h] = (_Float16*)carve(16 * 128 * 2);
  }
  // f32 biases
  float* b_init0 = (float*)carve(128 * 4);
  float* b_init1 = (float*)carve(256 * 4);
  float* b_init2 = (float*)carve(256 * 4);
  float* imu_bs[2] = {(float*)carve(1024 * 4), (float*)carve(1024 * 4)};
  float* ang_bs[2] = {(float*)carve(1024 * 4), (float*)carve(1024 * 4)};
  float* hbias0[12]; float* hbias1[12]; float* hbias2[12];
  for (int h = 0; h < 12; h++) {
    hbias0[h] = (float*)carve(256 * 4);
    hbias1[h] = (float*)carve(128 * 4);
    hbias2[h] = (float*)carve(16 * 4);
  }
  // activations
  _Float16* xtpack = (_Float16*)carve((size_t)BATCH * 32 * 2);
  _Float16* mlp1 = (_Float16*)carve((size_t)BATCH * 128 * 2);
  _Float16* mlp2 = (_Float16*)carve((size_t)BATCH * 256 * 2);
  _Float16* feat_t = (_Float16*)carve((size_t)BATCH * 512 * 2);
  _Float16* feat_r = (_Float16*)carve((size_t)BATCH * 512 * 2);
  _Float16* ab1 = (_Float16*)carve((size_t)BATCH * 256 * 2);
  _Float16* ab2 = (_Float16*)carve((size_t)BATCH * 128 * 2);
  float* headout = (float*)carve((size_t)12 * BATCH * 16 * 4);
  float* xt_prev = (float*)carve((size_t)BATCH * 6 * 4);
  float* P_prev = (float*)carve((size_t)BATCH * 36 * 4);

  auto cvtW = [&](const void* src, _Float16* dst, int N, int K, int Np, int Kp) {
    int tot = Np * Kp;
    cvt_pad<<<(tot + 255) / 256, 256, 0, stream>>>((const float*)src, dst, N, K,
                                                   Np, Kp);
  };
  auto cvtB = [&](const void* src, float* dst, int N, int Np) {
    bias_pad<<<(Np + 255) / 256, 256, 0, stream>>>((const float*)src, dst, N, Np);
  };
  auto sumB = [&](const void* a, const void* b, float* dst, int n) {
    bias_sum<<<(n + 255) / 256, 256, 0, stream>>>((const float*)a,
                                                  (const float*)b, dst, n);
  };
  // weight prep (runs in-graph every call; deterministic)
  cvtW(d_in[INIT_BASE + 0], w_init0, 128, 6, 128, 32);
  cvtB(d_in[INIT_BASE + 1], b_init0, 128, 128);
  cvtW(d_in[INIT_BASE + 2], w_init1, 256, 128, 256, 128);
  cvtB(d_in[INIT_BASE + 3], b_init1, 256, 256);
  cvtW(d_in[INIT_BASE + 4], w_init2, 256, 256, 256, 256);
  cvtB(d_in[INIT_BASE + 5], b_init2, 256, 256);
  cvtW(d_in[IMU_BASE + 0], imu_w[0], 1024, 6, 1024, 32);
  cvtW(d_in[IMU_BASE + 1], imu_w[1], 1024, 256, 1024, 256);
  sumB(d_in[IMU_BASE + 2], d_in[IMU_BASE + 3], imu_bs[0], 1024);
  cvtW(d_in[IMU_BASE + 4], imu_w[2], 1024, 256, 1024, 256);
  cvtW(d_in[IMU_BASE + 5], imu_w[3], 1024, 256, 1024, 256);
  sumB(d_in[IMU_BASE + 6], d_in[IMU_BASE + 7], imu_bs[1], 1024);
  cvtW(d_in[ANG_BASE + 0], ang_w[0], 1024, 3, 1024, 32);
  cvtW(d_in[ANG_BASE + 1], ang_w[1], 1024, 256, 1024, 256);
  sumB(d_in[ANG_BASE + 2], d_in[ANG_BASE + 3], ang_bs[0], 1024);
  cvtW(d_in[ANG_BASE + 4], ang_w[2], 1024, 256, 1024, 256);
  cvtW(d_in[ANG_BASE + 5], ang_w[3], 1024, 256, 1024, 256);
  sumB(d_in[ANG_BASE + 6], d_in[ANG_BASE + 7], ang_bs[1], 1024);
  for (int ty = 0; ty < 4; ty++)
    for (int br = 0; br < 3; br++) {
      int h = ty * 3 + br, bi = typeBase[ty][br];
      cvtW(d_in[bi + 0], hw0[h], 256, 512, 256, 512);
      cvtB(d_in[bi + 1], hbias0[h], 256, 256);
      cvtW(d_in[bi + 2], hw1[h], 128, 256, 128, 256);
      cvtB(d_in[bi + 3], hbias1[h], 128, 128);
      cvtW(d_in[bi + 4], hw2[h], headN[ty], 128, 16, 128);
      cvtB(d_in[bi + 5], hbias2[h], headN[ty], 16);
    }
  init_state<<<(BATCH * 36 + 255) / 256, 256, 0, stream>>>(xt_prev, P_prev);

  for (int t = 0; t < TT; t++) {
    pack_xt_k<<<(BATCH * 32 + 255) / 256, 256, 0, stream>>>(xt_prev, xtpack);
    launch_gemm(stream, xtpack, 32, w_init0, 32, b_init0, BATCH, 32, 128, mlp1,
                128, nullptr, 0, 1);
    launch_gemm(stream, mlp1, 128, w_init1, 128, b_init1, BATCH, 128, 256, mlp2,
                256, nullptr, 0, 1);
    launch_gemm(stream, mlp2, 256, w_init2, 256, b_init2, BATCH, 256, 256,
                feat_t, 512, nullptr, 0, 0);
    copy_x0<<<(BATCH * 256 + 255) / 256, 256, 0, stream>>>(feat_t, feat_r);
    lstm2<<<BATCH / 16, 256, LSTM_SMEM, stream>>>(
        acc, ang, t, 1, imu_w[0], imu_w[1], imu_w[2], imu_w[3], imu_bs[0],
        imu_bs[1], feat_t);
    lstm2<<<BATCH / 16, 256, LSTM_SMEM, stream>>>(
        acc, ang, t, 0, ang_w[0], ang_w[1], ang_w[2], ang_w[3], ang_bs[0],
        ang_bs[1], feat_r);
    for (int ty = 0; ty < 4; ty++)
      for (int br = 0; br < 3; br++) {
        int h = ty * 3 + br;
        const _Float16* fin = (ty == 1) ? feat_r : feat_t;
        launch_gemm(stream, fin, 512, hw0[h], 512, hbias0[h], BATCH, 512, 256,
                    ab1, 256, nullptr, 0, 1);
        launch_gemm(stream, ab1, 256, hw1[h], 256, hbias1[h], BATCH, 256, 128,
                    ab2, 128, nullptr, 0, 1);
        launch_gemm(stream, ab2, 128, hw2[h], 128, hbias2[h], BATCH, 128, 16,
                    nullptr, 0, headout + (size_t)h * BATCH * 16, 16, 0);
      }
    kalman<<<BATCH / 256, 256, 0, stream>>>(t, headout, cw, zt, Rm, acc, ang,
                                            xt_prev, P_prev, (float*)d_out);
  }
}
